// MixtureOfExperts_81930796138861
// MI455X (gfx1250) — matrix operand order repaired and checked by reading, not compiled, not run on hardware
//
#include <hip/hip_runtime.h>
#include <hip/hip_bf16.h>

#define N_EXPERTS 8
#define TOPK 2
#define D_MODEL 1024
#define N_TOK 8192

typedef __bf16 bf16_t;
typedef bf16_t v4bf  __attribute__((ext_vector_type(4)));
typedef bf16_t v8bf  __attribute__((ext_vector_type(8)));
typedef bf16_t v16bf __attribute__((ext_vector_type(16)));
typedef float  v8f   __attribute__((ext_vector_type(8)));
typedef int    v4i   __attribute__((ext_vector_type(4)));

// exact pointer types for the async global->LDS builtin (param 0 is
// "int __vector(4) __device__ *" per the compiler diagnostic)
typedef __attribute__((address_space(1))) v4i* gptr_b128;
typedef __attribute__((address_space(3))) v4i* lptr_b128;

union ABFrag { v16bf v; v8bf h[2]; };

#if defined(__has_builtin)
#if __has_builtin(__builtin_amdgcn_global_load_async_to_lds_b128) && \
    __has_builtin(__builtin_amdgcn_s_wait_asynccnt)
#define USE_ASYNC_LDS 1
#endif
#endif

// ---------------------------------------------------------------------------
// 0) zero the output and the per-expert counters
// ---------------------------------------------------------------------------
__global__ __launch_bounds__(256) void moe_zero_kernel(float4* __restrict__ out4,
                                                       int n4, int* __restrict__ counts) {
    int i = blockIdx.x * blockDim.x + threadIdx.x;
    if (i < n4) out4[i] = make_float4(0.f, 0.f, 0.f, 0.f);
    if (i < N_EXPERTS) counts[i] = 0;
}

// ---------------------------------------------------------------------------
// 1) x (fp32) -> x_bf (bf16), 4 elements / thread
// ---------------------------------------------------------------------------
__global__ __launch_bounds__(256) void moe_cvt_x_kernel(const float4* __restrict__ x4,
                                                        bf16_t* __restrict__ xb, int n4) {
    int i = blockIdx.x * blockDim.x + threadIdx.x;
    if (i >= n4) return;
    float4 v = x4[i];
    v4bf o = { (bf16_t)v.x, (bf16_t)v.y, (bf16_t)v.z, (bf16_t)v.w };
    *(v4bf*)(xb + (size_t)i * 4) = o;
}

// ---------------------------------------------------------------------------
// 2) W[e][d][h] fp32 -> Wb[e][h][d] bf16 (column-major for WMMA B fragments)
//    grid (D/32, D/32, E), block (32,8)
// ---------------------------------------------------------------------------
__global__ __launch_bounds__(256) void moe_cvt_w_kernel(const float* __restrict__ W,
                                                        bf16_t* __restrict__ Wb) {
    __shared__ float tile[32][33];
    const int e  = blockIdx.z;
    const int h0 = blockIdx.x * 32;
    const int d0 = blockIdx.y * 32;
    const float* src = W + (size_t)e * D_MODEL * D_MODEL;
    bf16_t*      dst = Wb + (size_t)e * D_MODEL * D_MODEL;
#pragma unroll
    for (int r = threadIdx.y; r < 32; r += 8)
        tile[r][threadIdx.x] = src[(size_t)(d0 + r) * D_MODEL + h0 + threadIdx.x];
    __syncthreads();
#pragma unroll
    for (int r = threadIdx.y; r < 32; r += 8)
        dst[(size_t)(h0 + r) * D_MODEL + d0 + threadIdx.x] = (bf16_t)tile[threadIdx.x][r];
}

// ---------------------------------------------------------------------------
// 3) router: one wave32 per token; logits = x @ Wr + br; top-2 + softmax;
//    append (token, gate) to the chosen experts' lists.
// ---------------------------------------------------------------------------
__global__ __launch_bounds__(256) void moe_router_kernel(const float* __restrict__ x,
                                                         const float* __restrict__ Wr,
                                                         const float* __restrict__ br,
                                                         int* __restrict__ counts,
                                                         int* __restrict__ lists,
                                                         float* __restrict__ gates) {
    const int lane = threadIdx.x & 31;
    const int wave = threadIdx.x >> 5;
    const int tok  = blockIdx.x * (blockDim.x >> 5) + wave;
    if (tok >= N_TOK) return;

    float acc[N_EXPERTS];
#pragma unroll
    for (int e = 0; e < N_EXPERTS; e++) acc[e] = 0.f;

    const float* xt = x + (size_t)tok * D_MODEL;
#pragma unroll 4
    for (int i = 0; i < D_MODEL / 32; i++) {
        int d = i * 32 + lane;
        float xv = xt[d];
        const float4* wr4 = (const float4*)(Wr + (size_t)d * N_EXPERTS);
        float4 w0 = wr4[0], w1 = wr4[1];
        acc[0] += xv * w0.x; acc[1] += xv * w0.y;
        acc[2] += xv * w0.z; acc[3] += xv * w0.w;
        acc[4] += xv * w1.x; acc[5] += xv * w1.y;
        acc[6] += xv * w1.z; acc[7] += xv * w1.w;
    }
#pragma unroll
    for (int e = 0; e < N_EXPERTS; e++) {
#pragma unroll
        for (int m = 16; m >= 1; m >>= 1)
            acc[e] += __shfl_xor(acc[e], m, 32);
    }

    if (lane == 0) {
        float lg[N_EXPERTS];
#pragma unroll
        for (int e = 0; e < N_EXPERTS; e++) lg[e] = acc[e] + br[e];
        int i0 = 0;
#pragma unroll
        for (int e = 1; e < N_EXPERTS; e++) if (lg[e] > lg[i0]) i0 = e;
        int i1 = -1;
#pragma unroll
        for (int e = 0; e < N_EXPERTS; e++)
            if (e != i0 && (i1 < 0 || lg[e] > lg[i1])) i1 = e;
        float v0 = lg[i0], v1 = lg[i1];
        float eb = __expf(v1 - v0);          // v0 >= v1
        float inv = 1.f / (1.f + eb);
        float g0 = inv, g1 = eb * inv;

        int s0 = atomicAdd(&counts[i0], 1);
        lists[i0 * N_TOK + s0] = tok;
        gates[i0 * N_TOK + s0] = g0;
        int s1 = atomicAdd(&counts[i1], 1);
        lists[i1 * N_TOK + s1] = tok;
        gates[i1 * N_TOK + s1] = g1;
    }
}

// ---------------------------------------------------------------------------
// 4) grouped expert GEMM via bf16 WMMA, fp32 accumulate.
//    grid (Ntiles=8, Mtiles=128, E=8), 256 threads = 8 waves.
//    Block tile: 64 gathered tokens x 128 output cols, K=1024.
//    Each wave: 4 accumulators (M=64) sharing one B fragment per K-step.
//
//    LDS layout (interleaved j-tiles so every DS offset is a small immediate):
//      physical token-row r (0..63) lives at byte offset
//        (r % 16) * GROUP_B + (r / 16) * ROW_B
//      GROUP_B = 4*2048 + 16 (pad per 4-row group -> conflict-free banks:
//      group stride = 2052 words == 4 mod 64), ROW_B = 2048 (16B-aligned).
//      Max DS immediate = 3*2048 + 2046 + 16 < 8.2 KB.
// ---------------------------------------------------------------------------
#define TILE_M 64
#define ROW_B   2048                  // bytes per token row (1024 bf16)
#define GROUP_B (4 * ROW_B + 16)      // 8208 bytes per r16-group of 4 rows

__global__ __launch_bounds__(256) void moe_gemm_kernel(const bf16_t* __restrict__ xb,
                                                       const bf16_t* __restrict__ Wb,
                                                       const float* __restrict__ bias,
                                                       const int* __restrict__ counts,
                                                       const int* __restrict__ lists,
                                                       const float* __restrict__ gates,
                                                       float* __restrict__ out) {
    const int e     = blockIdx.z;
    const int m0    = blockIdx.y * TILE_M;
    const int cnt   = counts[e];
    if (m0 >= cnt) return;
    const int valid = min(TILE_M, cnt - m0);

    __shared__ __align__(16) unsigned char smem[16 * GROUP_B];  // 131328 B

    const int* lst = lists + e * N_TOK + m0;
    // cooperative gather of 64 token rows (bf16) into LDS, 16B chunks / thread
#ifdef USE_ASYNC_LDS
    for (int c = threadIdx.x; c < TILE_M * (D_MODEL / 8); c += blockDim.x) {
        int row = c >> 7;                 // 128 chunks per row
        int off = (c & 127) * 16;         // byte offset within row
        int tok = lst[min(row, valid - 1)];
        __builtin_amdgcn_global_load_async_to_lds_b128(
            (gptr_b128)((const char*)xb + (size_t)tok * ROW_B + off),
            (lptr_b128)(smem + (row & 15) * GROUP_B + (row >> 4) * ROW_B + off),
            0, 0);
    }
    __builtin_amdgcn_s_wait_asynccnt(0);
#else
    for (int c = threadIdx.x; c < TILE_M * (D_MODEL / 8); c += blockDim.x) {
        int row = c >> 7;
        int off = (c & 127) * 16;
        int tok = lst[min(row, valid - 1)];
        *(v8bf*)(smem + (row & 15) * GROUP_B + (row >> 4) * ROW_B + off) =
            *(const v8bf*)((const char*)xb + (size_t)tok * ROW_B + off);
    }
#endif
    __syncthreads();

    const int wave  = threadIdx.x >> 5;
    const int lane  = threadIdx.x & 31;
    const int half  = lane >> 4;          // K-half selector
    const int col   = blockIdx.x * 128 + wave * 16 + (lane & 15);

    const bf16_t* bcol = Wb + ((size_t)e * D_MODEL + col) * D_MODEL;
    // single A base per lane; all tile/K selection via DS offset immediates
    const unsigned char* abase = smem + (lane & 15) * GROUP_B;

    v8f acc[4] = {v8f{}, v8f{}, v8f{}, v8f{}};
#pragma unroll 4
    for (int k0 = 0; k0 < D_MODEL; k0 += 32) {
        // B 32x16: lane = col; lanes 0-15 K {0..7,16..23}, lanes 16-31 K {8..15,24..31}
        ABFrag b;
        b.h[0] = *(const v8bf*)&bcol[k0 + half * 8];
        b.h[1] = *(const v8bf*)&bcol[k0 + half * 8 + 16];
        // A 16x32: lanes 0-15 K {0..7,16..23}, lanes 16-31 K {8..15,24..31}
        const int ka = 2 * (k0 + half * 8);       // byte offset of low half
        ABFrag a0, a1, a2, a3;
        a0.h[0] = *(const v8bf*)(abase + 0 * ROW_B + ka);
        a0.h[1] = *(const v8bf*)(abase + 0 * ROW_B + ka + 32);
        a1.h[0] = *(const v8bf*)(abase + 1 * ROW_B + ka);
        a1.h[1] = *(const v8bf*)(abase + 1 * ROW_B + ka + 32);
        a2.h[0] = *(const v8bf*)(abase + 2 * ROW_B + ka);
        a2.h[1] = *(const v8bf*)(abase + 2 * ROW_B + ka + 32);
        a3.h[0] = *(const v8bf*)(abase + 3 * ROW_B + ka);
        a3.h[1] = *(const v8bf*)(abase + 3 * ROW_B + ka + 32);

        acc[0] = __builtin_amdgcn_wmma_f32_16x16x32_bf16(false, a0.v, false, b.v,
                                                         (short)0, acc[0], false, false);
        acc[1] = __builtin_amdgcn_wmma_f32_16x16x32_bf16(false, a1.v, false, b.v,
                                                         (short)0, acc[1], false, false);
        acc[2] = __builtin_amdgcn_wmma_f32_16x16x32_bf16(false, a2.v, false, b.v,
                                                         (short)0, acc[2], false, false);
        acc[3] = __builtin_amdgcn_wmma_f32_16x16x32_bf16(false, a3.v, false, b.v,
                                                         (short)0, acc[3], false, false);
    }

    // D layout: lane element i -> M = half*8 + i (within 16-row tile), N = lane&15
    const float bcv = bias[e * D_MODEL + col];
    const float* gts = gates + e * N_TOK + m0;
#pragma unroll
    for (int j = 0; j < 4; j++) {
#pragma unroll
        for (int i = 0; i < 8; i++) {
            int m = j * 16 + half * 8 + i;
            if (m < valid) {
                int tok = lst[m];
                atomicAdd(&out[(size_t)tok * D_MODEL + col], gts[m] * (acc[j][i] + bcv));
            }
        }
    }
}

// ---------------------------------------------------------------------------
extern "C" void kernel_launch(void* const* d_in, const int* in_sizes, int n_in,
                              void* d_out, int out_size, void* d_ws, size_t ws_size,
                              hipStream_t stream) {
    const float* x  = (const float*)d_in[0];   // [N, d]
    const float* Wr = (const float*)d_in[1];   // [d, E]
    const float* br = (const float*)d_in[2];   // [E]
    const float* W  = (const float*)d_in[3];   // [E, d, d]
    const float* b  = (const float*)d_in[4];   // [E, d]
    float* out = (float*)d_out;                // [N, d]

    char* ws = (char*)d_ws;
    bf16_t* xb = (bf16_t*)ws;
    ws += (size_t)N_TOK * D_MODEL * sizeof(bf16_t);               // 16 MB
    bf16_t* Wb = (bf16_t*)ws;
    ws += (size_t)N_EXPERTS * D_MODEL * D_MODEL * sizeof(bf16_t); // 16 MB
    int* counts = (int*)ws;  ws += 128;
    int* lists  = (int*)ws;  ws += (size_t)N_EXPERTS * N_TOK * sizeof(int);   // 256 KB
    float* gates = (float*)ws;                                                // 256 KB

    const int n4 = N_TOK * D_MODEL / 4;                           // 2M float4
    moe_zero_kernel<<<(n4 + 255) / 256, 256, 0, stream>>>((float4*)out, n4, counts);
    moe_cvt_x_kernel<<<(n4 + 255) / 256, 256, 0, stream>>>((const float4*)x, xb, n4);
    moe_cvt_w_kernel<<<dim3(D_MODEL / 32, D_MODEL / 32, N_EXPERTS),
                       dim3(32, 8), 0, stream>>>(W, Wb);
    moe_router_kernel<<<N_TOK / 8, 256, 0, stream>>>(x, Wr, br, counts, lists, gates);
    moe_gemm_kernel<<<dim3(D_MODEL / 128, N_TOK / TILE_M, N_EXPERTS),
                      256, 0, stream>>>(xb, Wb, b, counts, lists, gates, out);
}
